// BigBirdText_70360154243241
// MI455X (gfx1250) — matrix operand order repaired and checked by reading, not compiled, not run on hardware
//
#include <hip/hip_runtime.h>
#include <hip/hip_bf16.h>
#include <stdint.h>

// ---------------------------------------------------------------------------
// BigBird encoder forward for MI455X (gfx1250, wave32, WMMA bf16 path).
// All matmuls use v_wmma_f32_16x16x32_bf16 with fp32 accumulation.
// Weights are pre-transposed to [N][K] bf16 so every WMMA B fragment is two
// contiguous ds_load_b128's; tile staging uses async global->LDS copies with
// double buffering (GLOBAL_LOAD_ASYNC_TO_LDS_B128 + s_wait_asynccnt).
// ---------------------------------------------------------------------------

typedef unsigned short ushort_t;
typedef unsigned int   uint_t;
typedef int v4i_t __attribute__((vector_size(16)));   // matches builtin proto

typedef __attribute__((ext_vector_type(16))) __bf16 v16bf;
typedef __attribute__((ext_vector_type(8)))  float  v8f;

// Model dimensions (match the reference).
constexpr int Bc  = 2;
constexpr int Sc  = 4096;
constexpr int Dc  = 1024;
constexpr int Hc  = 16;
constexpr int BSc = 64;
constexpr int Lc  = 4;
constexpr int FFc = 4096;
constexpr int HDc = Dc / Hc;       // 64
constexpr int NBc = Sc / BSc;      // 64
constexpr int NMc = NBc - 2;       // 62
constexpr int Rc  = 3;

__device__ __forceinline__ ushort_t f2bf(float f) {
  uint_t u = __float_as_uint(f);
  u += 0x7FFFu + ((u >> 16) & 1u);        // round-to-nearest-even
  return (ushort_t)(u >> 16);
}

__device__ __forceinline__ float gelu_approx(float x) {
  const float k0 = 0.7978845608028654f;   // sqrt(2/pi)
  const float k1 = 0.044715f;
  float x3 = x * x * x;
  return 0.5f * x * (1.0f + tanhf(k0 * (x + k1 * x3)));
}

union AFrag { v16bf v; uint_t u[8]; };
union CFrag { v8f   v; float  f[8]; };

__device__ __forceinline__ v8f wmma_bf16(v16bf a, v16bf b, v8f c) {
  return __builtin_amdgcn_wmma_f32_16x16x32_bf16(
      /*neg_a=*/false, a, /*neg_b=*/false, b,
      /*c_mod=*/(short)0, c, /*reuse_a=*/false, /*reuse_b=*/false);
}

// ---------------------------------------------------------------------------
// Async global -> LDS 16-byte copy (gfx1250 GLOBAL_LOAD_ASYNC_TO_LDS_B128),
// with a synchronous uint4 fallback if the builtin is not exposed.
// ---------------------------------------------------------------------------
#if defined(__has_builtin)
#  if __has_builtin(__builtin_amdgcn_global_load_async_to_lds_b128)
#    define HAVE_ASYNC_LDS 1
#  endif
#endif
#ifndef HAVE_ASYNC_LDS
#  define HAVE_ASYNC_LDS 0
#endif

__device__ __forceinline__ void cp16_g2s(ushort_t* lds, const ushort_t* g) {
#if HAVE_ASYNC_LDS
  __builtin_amdgcn_global_load_async_to_lds_b128(
      (__attribute__((address_space(1))) v4i_t*)g,
      (__attribute__((address_space(3))) v4i_t*)lds, 0, 0);
#else
  *(uint4*)lds = *(const uint4*)g;
#endif
}

__device__ __forceinline__ void wait_async_cp() {
#if HAVE_ASYNC_LDS
#  if __has_builtin(__builtin_amdgcn_s_wait_asynccnt)
  __builtin_amdgcn_s_wait_asynccnt(0);
#  else
  asm volatile("s_wait_asynccnt 0x0" ::: "memory");
#  endif
#endif
}

// Load a 16x32 bf16 A fragment from LDS (K contiguous per row, stride `lds`).
// ISA A layout: K = {2r,2r+1} + 8*laneHalf (+16 for r>=4).
__device__ __forceinline__ void load_afrag(AFrag& a, const ushort_t* base,
                                           int row, int lds, int ks,
                                           int am, int ah) {
  const ushort_t* ap = base + (row + am) * lds + ks + ah * 8;
  a.u[0] = *(const uint_t*)(ap + 0);  a.u[1] = *(const uint_t*)(ap + 2);
  a.u[2] = *(const uint_t*)(ap + 4);  a.u[3] = *(const uint_t*)(ap + 6);
  a.u[4] = *(const uint_t*)(ap + 16); a.u[5] = *(const uint_t*)(ap + 18);
  a.u[6] = *(const uint_t*)(ap + 20); a.u[7] = *(const uint_t*)(ap + 22);
}

// Load a 32x16 bf16 B fragment from LDS stored K-contiguous per column
// (i.e. [n][k] layout).  B layout matches A: K = {2r,2r+1} + 8*laneHalf
// (+16 for r>=4).
__device__ __forceinline__ void load_bfrag(AFrag& b, const ushort_t* base,
                                           int col, int lds, int ks,
                                           int bn, int bh) {
  const ushort_t* bp = base + (col + bn) * lds + ks + bh * 8;
#pragma unroll
  for (int r = 0; r < 8; ++r) b.u[r] = *(const uint_t*)(bp + 2 * r + (r >= 4 ? 8 : 0));
}

// ---------------------------------------------------------------------------
// GEMM: C[M,N] = A[M,K](bf16) * W[K,N] + bias, W supplied transposed [N][K].
// Block tile 128x64, BK=64, 8 waves each own a 32x32 patch (2x2 WMMA tiles).
// Double-buffered LDS with async staging.
// flags: bit0 = bf16 output, bit1 = gelu, bit2 = add fp32 residual.
// Requires M%128==0, N%64==0, K%64==0 (true for all shapes here).
// ---------------------------------------------------------------------------
constexpr int GBM = 128, GBN = 64, GBK = 64;
constexpr int GLA = GBK + 8;   // 72-ushort row stride (bank-conflict pad)
constexpr int GLB = GBK + 8;   // B tile is [n][k], K-stride padded likewise

__global__ __launch_bounds__(256) void gemm_bf16_kernel(
    const ushort_t* __restrict__ A, const ushort_t* __restrict__ Bt,
    const float* __restrict__ bias, const float* __restrict__ resid,
    void* __restrict__ Cout, int M, int N, int K, int flags)
{
  __shared__ __attribute__((aligned(16))) ushort_t sA[2][GBM * GLA];
  __shared__ __attribute__((aligned(16))) ushort_t sB[2][GBN * GLB];

  const int t = threadIdx.x;
  const int w = t >> 5, lane = t & 31;
  const int m0 = blockIdx.y * GBM, n0 = blockIdx.x * GBN;
  const int wm = w >> 1, wn = w & 1;          // wave patch origin /32
  const int am = lane & 15, ah = lane >> 4;   // A-frag row / K-half
  const int dn = lane & 15, dh = lane >> 4;   // B/D-frag col / half

  CFrag acc[2][2];
#pragma unroll
  for (int i = 0; i < 2; ++i)
#pragma unroll
    for (int j = 0; j < 2; ++j)
#pragma unroll
      for (int r = 0; r < 8; ++r) acc[i][j].f[r] = 0.f;

  // Stage one K-tile into LDS buffer `buf`.
  auto stage = [&](int buf, int k0) {
#pragma unroll
    for (int i = 0; i < 4; ++i) {           // A tile: 128x64 bf16
      int idx = (t + i * 256) * 8;
      int r = idx >> 6, c = idx & 63;
      cp16_g2s(&sA[buf][r * GLA + c], A + (size_t)(m0 + r) * K + k0 + c);
    }
#pragma unroll
    for (int i = 0; i < 2; ++i) {           // B tile: 64 cols x 64 k ([n][k])
      int idx = (t + i * 256) * 8;
      int r = idx >> 6, c = idx & 63;
      cp16_g2s(&sB[buf][r * GLB + c], Bt + (size_t)(n0 + r) * K + k0 + c);
    }
  };

  stage(0, 0);
  wait_async_cp();
  __syncthreads();

  int buf = 0;
  for (int k0 = 0; k0 < K; k0 += GBK) {
    if (k0 + GBK < K) stage(buf ^ 1, k0 + GBK);   // fill next while computing
    if (k0 + 2 * GBK < K)                         // and nudge tile after that
      __builtin_prefetch(A + (size_t)(m0 + (t >> 1)) * K + k0 + 2 * GBK, 0, 0);

#pragma unroll
    for (int ks = 0; ks < GBK; ks += 32) {
      AFrag a[2], b[2];
#pragma unroll
      for (int i = 0; i < 2; ++i)
        load_afrag(a[i], sA[buf], wm * 32 + i * 16, GLA, ks, am, ah);
#pragma unroll
      for (int j = 0; j < 2; ++j)
        load_bfrag(b[j], sB[buf], wn * 32 + j * 16, GLB, ks, dn, dh);
#pragma unroll
      for (int i = 0; i < 2; ++i)
#pragma unroll
        for (int j = 0; j < 2; ++j)
          acc[i][j].v = wmma_bf16(a[i].v, b[j].v, acc[i][j].v);
    }
    wait_async_cp();
    __syncthreads();
    buf ^= 1;
  }

  // Fused epilogue.
#pragma unroll
  for (int i = 0; i < 2; ++i) {
#pragma unroll
    for (int j = 0; j < 2; ++j) {
      const int col = n0 + wn * 32 + j * 16 + dn;
#pragma unroll
      for (int r = 0; r < 8; ++r) {
        const int row = m0 + wm * 32 + i * 16 + r + 8 * dh;
        float c = acc[i][j].f[r] + bias[col];
        if (flags & 2) c = gelu_approx(c);
        if (flags & 4) c += resid[(size_t)row * N + col];
        if (flags & 1) ((ushort_t*)Cout)[(size_t)row * N + col] = f2bf(c);
        else           ((float*)   Cout)[(size_t)row * N + col] = c;
      }
    }
  }
}

// ---------------------------------------------------------------------------
// BigBird attention, flash-style, one workgroup per (b, h, q-block).
// q-block 0 and NB-1 sweep all key blocks (== reference global attention);
// middle q-block n sweeps {n-1, n, n+1, 0, NB-1, rand[0..2]}.
// V is staged transposed so P*V B fragments are contiguous b128 loads.
// ---------------------------------------------------------------------------
__global__ __launch_bounds__(256) void bigbird_attn_kernel(
    const ushort_t* __restrict__ q, const ushort_t* __restrict__ k,
    const ushort_t* __restrict__ v, const float* __restrict__ neg,
    const int* __restrict__ rand_attn, ushort_t* __restrict__ ctx)
{
  constexpr int LQ = HDc + 8;   // 72-ushort stride
  constexpr int LS = BSc + 2;   // 66-float stride
  __shared__ __attribute__((aligned(16))) ushort_t q_s[BSc * LQ]; // [row][hd]
  __shared__ __attribute__((aligned(16))) ushort_t k_s[BSc * LQ]; // [key][hd]
  __shared__ __attribute__((aligned(16))) ushort_t v_t[HDc * LQ]; // [hd][key]
  __shared__ __attribute__((aligned(16))) ushort_t p_s[BSc * LQ]; // [row][key]
  __shared__ float    s_s[BSc * LS];
  __shared__ float    c_s[BSc * LS];
  __shared__ float    mrow[BSc], lrow[BSc], arow[BSc];
  __shared__ int      klist[NBc];

  const int n = blockIdx.x, h = blockIdx.y, b = blockIdx.z;
  const int t = threadIdx.x;
  const int w = t >> 5, lane = t & 31;
  const bool isGlobal = (n == 0) || (n == NBc - 1);
  const int nkey = isGlobal ? NBc : 8;

  if (isGlobal) {
    if (t < NBc) klist[t] = t;
  } else if (t < 8) {
    int kv;
    if      (t == 0) kv = n - 1;
    else if (t == 1) kv = n;
    else if (t == 2) kv = n + 1;
    else if (t == 3) kv = 0;
    else if (t == 4) kv = NBc - 1;
    else             kv = rand_attn[(h * NMc + (n - 1)) * Rc + (t - 5)];
    klist[t] = kv;
  }

  // Stage Q tile (async); init running stats + context accumulator.
#pragma unroll
  for (int i = 0; i < 2; ++i) {
    int idx = (t + i * 256) * 8;
    int r = idx >> 6, c = idx & 63;
    cp16_g2s(&q_s[r * LQ + c],
             q + ((size_t)(b * Sc + n * BSc + r)) * Dc + h * HDc + c);
  }
  for (int e = t; e < BSc * LS; e += 256) c_s[e] = 0.f;
  if (t < BSc) { mrow[t] = -1e30f; lrow[t] = 0.f; }

  const float scale = 0.125f;  // 1/sqrt(HD)
  const int am = lane & 15, ah = lane >> 4;
  const int dn = lane & 15, dh = lane >> 4;

  for (int kbi = 0; kbi < nkey; ++kbi) {
    const int kb = klist[kbi];

    // K tile: direct [key][hd] (async).  V tile: transposed to [hd][key].
#pragma unroll
    for (int i = 0; i < 2; ++i) {
      int idx = (t + i * 256) * 8;
      int r = idx >> 6, c = idx & 63;
      cp16_g2s(&k_s[r * LQ + c],
               k + ((size_t)(b * Sc + kb * BSc + r)) * Dc + h * HDc + c);
    }
    for (int e = t; e < BSc * HDc; e += 256) {
      int r = e >> 6, c = e & 63;   // r = key, c = hd (coalesced global read)
      v_t[c * LQ + r] = v[((size_t)(b * Sc + kb * BSc + r)) * Dc + h * HDc + c];
    }
    wait_async_cp();
    __syncthreads();

    // S = Q * K^T : wave (wm,wn) -> rows wm*16..+15, cols wn*32..+31.
    {
      const int wm = w >> 1, wn = w & 1;
      CFrag acc0, acc1;
#pragma unroll
      for (int r = 0; r < 8; ++r) { acc0.f[r] = 0.f; acc1.f[r] = 0.f; }
      for (int ks = 0; ks < HDc; ks += 32) {
        AFrag a;
        load_afrag(a, q_s, wm * 16, LQ, ks, am, ah);
#pragma unroll
        for (int j = 0; j < 2; ++j) {
          AFrag bf;   // B col = key row of k_s; K = hd contiguous
          load_bfrag(bf, k_s, wn * 32 + j * 16, LQ, ks, dn, dh);
          if (j == 0) acc0.v = wmma_bf16(a.v, bf.v, acc0.v);
          else        acc1.v = wmma_bf16(a.v, bf.v, acc1.v);
        }
      }
#pragma unroll
      for (int j = 0; j < 2; ++j) {
        const int col = wn * 32 + j * 16 + dn;
        const float m = neg[b * Sc + kb * BSc + col];
        const CFrag& ac = (j == 0) ? acc0 : acc1;
#pragma unroll
        for (int r = 0; r < 8; ++r)
          s_s[(wm * 16 + r + 8 * dh) * LS + col] = ac.f[r] * scale + m;
      }
    }
    __syncthreads();

    // Online softmax: waves 0-1 own one row each (no intra-wave divergence).
    if (t < BSc) {
      float bm = -1e30f;
      for (int c = 0; c < BSc; ++c) bm = fmaxf(bm, s_s[t * LS + c]);
      const float mn = fmaxf(mrow[t], bm);
      const float al = __expf(mrow[t] - mn);
      float sum = 0.f;
      for (int c = 0; c < BSc; ++c) {
        const float pv = __expf(s_s[t * LS + c] - mn);
        p_s[t * LQ + c] = f2bf(pv);
        sum += pv;
      }
      lrow[t] = lrow[t] * al + sum;
      mrow[t] = mn;
      arow[t] = al;
    }
    __syncthreads();
    for (int e = t; e < BSc * HDc; e += 256) {
      int r = e >> 6, c = e & 63;
      c_s[r * LS + c] *= arow[r];
    }
    __syncthreads();

    // CTX += P * V : 16 tiles over 8 waves, 2 tiles per wave.
    for (int ti = 0; ti < 2; ++ti) {
      const int tt = w + ti * 8;
      const int tm = tt >> 2, tn = tt & 3;
      CFrag acc;
#pragma unroll
      for (int r = 0; r < 8; ++r)
        acc.f[r] = c_s[(tm * 16 + r + 8 * dh) * LS + tn * 16 + dn];
      for (int ks = 0; ks < BSc; ks += 32) {
        AFrag a, bf;
        load_afrag(a, p_s, tm * 16, LQ, ks, am, ah);
        load_bfrag(bf, v_t, tn * 16, LQ, ks, dn, dh);  // [hd][key]: contiguous
        acc.v = wmma_bf16(a.v, bf.v, acc.v);
      }
#pragma unroll
      for (int r = 0; r < 8; ++r)
        c_s[(tm * 16 + r + 8 * dh) * LS + tn * 16 + dn] = acc.f[r];
    }
    __syncthreads();
  }

  // ctx / l -> [B,S,D] bf16 (head-major columns), ready for Wo GEMM.
  for (int e = t; e < BSc * HDc; e += 256) {
    int r = e >> 6, c = e & 63;
    float o = c_s[r * LS + c] / lrow[r];
    ctx[((size_t)(b * Sc + n * BSc + r)) * Dc + h * HDc + c] = f2bf(o);
  }
}

// ---------------------------------------------------------------------------
// Elementwise / reduction helper kernels.
// ---------------------------------------------------------------------------
__global__ void embed_kernel(const int* __restrict__ ids,
                             const float* __restrict__ emb,
                             float* __restrict__ x, int total) {
  int i = blockIdx.x * 256 + threadIdx.x;
  if (i >= total) return;
  int tok = i / Dc, d = i - tok * Dc;
  x[i] = emb[(size_t)ids[tok] * Dc + d];
}

__global__ void neg_mask_kernel(const int* __restrict__ mask,
                                float* __restrict__ neg, int total) {
  int i = blockIdx.x * 256 + threadIdx.x;
  if (i < total) neg[i] = (1.f - (float)mask[i]) * -1e9f;
}

// Transpose + fp32->bf16: in [Kd][Nd] fp32 (layer blockIdx.z) -> out [Nd][Kd].
__global__ __launch_bounds__(256) void transpose_cvt_kernel(
    const float* __restrict__ in, ushort_t* __restrict__ out, int Kd, int Nd)
{
  __shared__ float tile[64][65];
  const int t = threadIdx.x;
  const int kb = blockIdx.x * 64, nb = blockIdx.y * 64;
  const size_t lofs = (size_t)blockIdx.z * Kd * Nd;
  for (int e = t; e < 64 * 64; e += 256) {
    int r = e >> 6, c = e & 63;
    tile[r][c] = in[lofs + (size_t)(kb + r) * Nd + nb + c];
  }
  __syncthreads();
  for (int e = t; e < 64 * 64; e += 256) {
    int r = e >> 6, c = e & 63;
    out[lofs + (size_t)(nb + r) * Kd + kb + c] = f2bf(tile[c][r]);
  }
}

__global__ __launch_bounds__(256) void layernorm_kernel(
    const float* __restrict__ x, const float* __restrict__ g,
    const float* __restrict__ bt, void* __restrict__ out, int outbf)
{
  __shared__ float red[256];
  const int row = blockIdx.x, t = threadIdx.x;
  const float* xr = x + (size_t)row * Dc;
  float ls = 0.f;
#pragma unroll
  for (int i = 0; i < Dc / 256; ++i) ls += xr[t + i * 256];
  red[t] = ls; __syncthreads();
  for (int s = 128; s > 0; s >>= 1) { if (t < s) red[t] += red[t + s]; __syncthreads(); }
  const float mu = red[0] * (1.f / Dc);
  __syncthreads();
  float lv = 0.f;
#pragma unroll
  for (int i = 0; i < Dc / 256; ++i) { float d = xr[t + i * 256] - mu; lv += d * d; }
  red[t] = lv; __syncthreads();
  for (int s = 128; s > 0; s >>= 1) { if (t < s) red[t] += red[t + s]; __syncthreads(); }
  const float rinv = rsqrtf(red[0] * (1.f / Dc) + 1e-5f);
#pragma unroll
  for (int i = 0; i < Dc / 256; ++i) {
    int c = t + i * 256;
    float y = (xr[c] - mu) * rinv * g[c] + bt[c];
    if (outbf) ((ushort_t*)out)[(size_t)row * Dc + c] = f2bf(y);
    else       ((float*)   out)[(size_t)row * Dc + c] = y;
  }
}

__global__ void mean_rows_kernel(const float* __restrict__ xf,
                                 float* __restrict__ feats) {
  int i = blockIdx.x * 256 + threadIdx.x;   // b*Dc + d
  int b = i / Dc, d = i - b * Dc;
  const float* base = xf + (size_t)b * Sc * Dc + d;
  float s = 0.f;
  for (int r = 0; r < Sc; ++r) s += base[(size_t)r * Dc];
  feats[i] = s * (1.f / Sc);
}

__global__ __launch_bounds__(256) void l2norm_kernel(float* __restrict__ feats) {
  __shared__ float red[256];
  const int b = blockIdx.x, t = threadIdx.x;
  float* f = feats + (size_t)b * Dc;
  float s = 0.f;
#pragma unroll
  for (int i = 0; i < Dc / 256; ++i) { float vv = f[t + i * 256]; s += vv * vv; }
  red[t] = s; __syncthreads();
  for (int st = 128; st > 0; st >>= 1) { if (t < st) red[t] += red[t + st]; __syncthreads(); }
  const float inv = 1.f / fmaxf(sqrtf(red[0]), 1e-12f);
#pragma unroll
  for (int i = 0; i < Dc / 256; ++i) f[t + i * 256] *= inv;
}

__global__ void proj_kernel(const float* __restrict__ feats,
                            const float* __restrict__ W,
                            const float* __restrict__ bias,
                            float* __restrict__ out) {
  int i = blockIdx.x * 256 + threadIdx.x;   // b*512 + o
  int b = i >> 9, o = i & 511;
  const float* f = feats + (size_t)b * Dc;
  float s = bias[o];
  for (int d = 0; d < Dc; ++d) s += f[d] * W[(size_t)d * 512 + o];
  out[i] = s;
}

// ---------------------------------------------------------------------------
// Host driver.  Workspace use: ~319 MB (bf16 weights 100.7 MB, activations).
// ---------------------------------------------------------------------------
extern "C" void kernel_launch(void* const* d_in, const int* in_sizes, int n_in,
                              void* d_out, int out_size, void* d_ws, size_t ws_size,
                              hipStream_t stream) {
  (void)in_sizes; (void)n_in; (void)out_size; (void)ws_size;

  const int*   ids   = (const int*)  d_in[0];
  const int*   amask = (const int*)  d_in[1];
  const int*   randa = (const int*)  d_in[2];
  const float* emb   = (const float*)d_in[3];
  const float* Wq    = (const float*)d_in[4];
  const float* bq    = (const float*)d_in[5];
  const float* Wk    = (const float*)d_in[6];
  const float* bk    = (const float*)d_in[7];
  const float* Wv    = (const float*)d_in[8];
  const float* bv    = (const float*)d_in[9];
  const float* Wo    = (const float*)d_in[10];
  const float* bo    = (const float*)d_in[11];
  const float* ln1g  = (const float*)d_in[12];
  const float* ln1b  = (const float*)d_in[13];
  const float* W1    = (const float*)d_in[14];
  const float* b1    = (const float*)d_in[15];
  const float* W2    = (const float*)d_in[16];
  const float* b2    = (const float*)d_in[17];
  const float* ln2g  = (const float*)d_in[18];
  const float* ln2b  = (const float*)d_in[19];
  const float* lnfg  = (const float*)d_in[20];
  const float* lnfb  = (const float*)d_in[21];
  const float* projW = (const float*)d_in[22];
  const float* projb = (const float*)d_in[23];

  char* p = (char*)d_ws;
  auto alloc = [&](size_t bytes) {
    void* r = (void*)p;
    p += (bytes + 255) & ~(size_t)255;
    return r;
  };
  const size_t DD = (size_t)Dc * Dc, DF = (size_t)Dc * FFc;
  const size_t TOK = (size_t)Bc * Sc;           // 8192 tokens

  // All weight copies are transposed bf16: [N][K] per layer.
  ushort_t* Wq_b = (ushort_t*)alloc(Lc * DD * 2);
  ushort_t* Wk_b = (ushort_t*)alloc(Lc * DD * 2);
  ushort_t* Wv_b = (ushort_t*)alloc(Lc * DD * 2);
  ushort_t* Wo_b = (ushort_t*)alloc(Lc * DD * 2);
  ushort_t* W1_b = (ushort_t*)alloc(Lc * DF * 2);
  ushort_t* W2_b = (ushort_t*)alloc(Lc * DF * 2);
  float*    x    = (float*)   alloc(TOK * Dc * 4);
  float*    xf   = (float*)   alloc(TOK * Dc * 4);
  ushort_t* xn   = (ushort_t*)alloc(TOK * Dc * 2);
  ushort_t* qb   = (ushort_t*)alloc(TOK * Dc * 2);
  ushort_t* kb   = (ushort_t*)alloc(TOK * Dc * 2);
  ushort_t* vb   = (ushort_t*)alloc(TOK * Dc * 2);
  ushort_t* ctxb = (ushort_t*)alloc(TOK * Dc * 2);
  ushort_t* ffh  = (ushort_t*)alloc(TOK * FFc * 2);
  float*    neg  = (float*)   alloc(TOK * 4);
  float*    feats= (float*)   alloc((size_t)Bc * Dc * 4);

  dim3 blk(256);

  // Weights fp32 -> bf16 transposed (once per call; inputs never mutated).
  {
    dim3 gDD(Dc / 64, Dc / 64, Lc);
    transpose_cvt_kernel<<<gDD, blk, 0, stream>>>(Wq, Wq_b, Dc, Dc);
    transpose_cvt_kernel<<<gDD, blk, 0, stream>>>(Wk, Wk_b, Dc, Dc);
    transpose_cvt_kernel<<<gDD, blk, 0, stream>>>(Wv, Wv_b, Dc, Dc);
    transpose_cvt_kernel<<<gDD, blk, 0, stream>>>(Wo, Wo_b, Dc, Dc);
    dim3 gW1(Dc / 64, FFc / 64, Lc);   // W1: [Dc][FFc]
    transpose_cvt_kernel<<<gW1, blk, 0, stream>>>(W1, W1_b, Dc, FFc);
    dim3 gW2(FFc / 64, Dc / 64, Lc);   // W2: [FFc][Dc]
    transpose_cvt_kernel<<<gW2, blk, 0, stream>>>(W2, W2_b, FFc, Dc);
  }

  embed_kernel<<<(int)(TOK * Dc / 256), blk, 0, stream>>>(ids, emb, x, (int)(TOK * Dc));
  neg_mask_kernel<<<(int)((TOK + 255) / 256), blk, 0, stream>>>(amask, neg, (int)TOK);

  const int  Mtok = (int)TOK;                     // 8192
  const dim3 gD(Dc / GBN, Mtok / GBM);            // (16, 64) for N=1024
  const dim3 gF(FFc / GBN, Mtok / GBM);           // (64, 64) for N=4096
  const dim3 gA(NBc, Hc, Bc);

  for (int l = 0; l < Lc; ++l) {
    layernorm_kernel<<<Mtok, blk, 0, stream>>>(x, ln1g + l * Dc, ln1b + l * Dc, xn, 1);
    gemm_bf16_kernel<<<gD, blk, 0, stream>>>(xn, Wq_b + l * DD, bq + l * Dc, nullptr, qb, Mtok, Dc, Dc, 1);
    gemm_bf16_kernel<<<gD, blk, 0, stream>>>(xn, Wk_b + l * DD, bk + l * Dc, nullptr, kb, Mtok, Dc, Dc, 1);
    gemm_bf16_kernel<<<gD, blk, 0, stream>>>(xn, Wv_b + l * DD, bv + l * Dc, nullptr, vb, Mtok, Dc, Dc, 1);
    bigbird_attn_kernel<<<gA, blk, 0, stream>>>(qb, kb, vb, neg, randa, ctxb);
    gemm_bf16_kernel<<<gD, blk, 0, stream>>>(ctxb, Wo_b + l * DD, bo + l * Dc, x, x, Mtok, Dc, Dc, 4);
    layernorm_kernel<<<Mtok, blk, 0, stream>>>(x, ln2g + l * Dc, ln2b + l * Dc, xn, 1);
    gemm_bf16_kernel<<<gF, blk, 0, stream>>>(xn, W1_b + l * DF, b1 + l * FFc, nullptr, ffh, Mtok, FFc, Dc, 1 | 2);
    gemm_bf16_kernel<<<gD, blk, 0, stream>>>(ffh, W2_b + l * DF, b2 + l * Dc, x, x, Mtok, Dc, FFc, 4);
  }

  layernorm_kernel<<<Mtok, blk, 0, stream>>>(x, lnfg, lnfb, xf, 0);
  mean_rows_kernel<<<(int)(Bc * Dc / 256), blk, 0, stream>>>(xf, feats);
  l2norm_kernel<<<Bc, blk, 0, stream>>>(feats);
  proj_kernel<<<(Bc * 512) / 256, blk, 0, stream>>>(feats, projW, projb, (float*)d_out);
}
